// MultiHeadAttentionClass_15857019257097
// MI455X (gfx1250) — matrix operand order repaired and checked by reading, not compiled, not run on hardware
//
#include <hip/hip_runtime.h>

typedef __bf16 bf16_t;
typedef __attribute__((ext_vector_type(16))) __bf16 v16bf;
typedef __attribute__((ext_vector_type(8)))  float  v8f;

#define HH   8
#define BBATCH 16
#define LLEN 1024
#define DMODEL 512
#define DKV  64
#define MTOT (BBATCH * LLEN)   /* 16384 */

union ABu { v16bf v; float4 q[2]; };

// A-matrix 16x32 bf16 tile (M x K). base points at [row0, k0]; lda in elements.
// Contiguous-k layout (matches load_b32x16): lane<16 -> M=lane, K = k0..k0+15;
//             lane>=16 -> M=lane-16, K = k0+16..k0+31.
__device__ __forceinline__ v16bf load_a16x32(const bf16_t* base, int lda, int lane) {
  const int m  = lane & 15;
  const int kb = (lane >> 4) << 4;        // 0 or 16
  const bf16_t* p = base + m * lda + kb;
  ABu u;
  u.q[0] = *(const float4*)(p);           // elements 0..7
  u.q[1] = *(const float4*)(p + 8);       // elements 8..15
  return u.v;
}

// B-matrix 32x16 bf16 tile (K x N), stored column-contiguous: memory is [n][k] row-major.
// base points at [n0, k0]; ldb in elements. lane<16 -> N=lane, K=k0..k0+15;
// lane>=16 -> N=lane-16, K=k0+16..k0+31 (16 contiguous elements per lane).
__device__ __forceinline__ v16bf load_b32x16(const bf16_t* base, int ldb, int lane) {
  const int n  = lane & 15;
  const int kb = (lane >> 4) << 4;        // 0 or 16
  const bf16_t* p = base + n * ldb + kb;
  ABu u;
  u.q[0] = *(const float4*)(p);
  u.q[1] = *(const float4*)(p + 8);
  return u.v;
}

__device__ __forceinline__ v8f wmma_bf16(v16bf a, v16bf b, v8f c) {
  return __builtin_amdgcn_wmma_f32_16x16x32_bf16(false, a, false, b, (short)0, c, false, false);
}

// ---------------------------------------------------------------- conversions
__global__ void cvt_f32_bf16(const float* __restrict__ src, bf16_t* __restrict__ dst, int n) {
  for (int i = blockIdx.x * blockDim.x + threadIdx.x; i < n; i += gridDim.x * blockDim.x)
    dst[i] = (bf16_t)src[i];
}

// w[h][d][e] f32 (H x 512 x 64) -> wt[h][e][d] bf16 (so B columns are contiguous in K)
__global__ void cvt_transpose_w(const float* __restrict__ src, bf16_t* __restrict__ dst) {
  const int n = HH * DMODEL * DKV;
  for (int i = blockIdx.x * blockDim.x + threadIdx.x; i < n; i += gridDim.x * blockDim.x) {
    int e = i & 63;
    int d = (i >> 6) & (DMODEL - 1);
    int h = i >> 15;   // 6+9 bits
    dst[((size_t)((h << 6) + e)) * DMODEL + d] = (bf16_t)src[i];
  }
}

// ------------------------------------------------- per-head Q/K/V projections
// out[h][m][e] = sum_d X[m][d] * W[h][d][e]   (M=16384, K=512, N=64)
// one wave computes a 16x64 output strip; grid = (256, 8 heads), block = 128.
// Register double-buffered K loop: next-iteration loads issue before current WMMAs.
__global__ __launch_bounds__(128) void qkv_gemm(
    const bf16_t* __restrict__ X,    // [16384][512]
    const bf16_t* __restrict__ Wt,   // [H][64][512]
    bf16_t* __restrict__ outB,       // [H][16384][64] bf16 or nullptr
    float*  __restrict__ outF)       // [H][16384][64] f32  or nullptr
{
  const int K = DMODEL;
  const int h    = blockIdx.y;
  const int wave = threadIdx.x >> 5;
  const int lane = threadIdx.x & 31;
  const int m0   = (blockIdx.x * 4 + wave) * 16;
  const bf16_t* Ab = X + (size_t)m0 * K;
  const bf16_t* Bb = Wt + (size_t)h * DKV * K;
  v8f c0 = {}, c1 = {}, c2 = {}, c3 = {};

  v16bf a  = load_a16x32(Ab, K, lane);
  v16bf b0 = load_b32x16(Bb + (size_t)0  * K, K, lane);
  v16bf b1 = load_b32x16(Bb + (size_t)16 * K, K, lane);
  v16bf b2 = load_b32x16(Bb + (size_t)32 * K, K, lane);
  v16bf b3 = load_b32x16(Bb + (size_t)48 * K, K, lane);
  for (int k0 = 0; k0 < K - 32; k0 += 32) {
    const int kn = k0 + 32;
    v16bf an = load_a16x32(Ab + kn, K, lane);
    v16bf n0 = load_b32x16(Bb + (size_t)0  * K + kn, K, lane);
    v16bf n1 = load_b32x16(Bb + (size_t)16 * K + kn, K, lane);
    v16bf n2 = load_b32x16(Bb + (size_t)32 * K + kn, K, lane);
    v16bf n3 = load_b32x16(Bb + (size_t)48 * K + kn, K, lane);
    c0 = wmma_bf16(a, b0, c0);
    c1 = wmma_bf16(a, b1, c1);
    c2 = wmma_bf16(a, b2, c2);
    c3 = wmma_bf16(a, b3, c3);
    a = an; b0 = n0; b1 = n1; b2 = n2; b3 = n3;
  }
  c0 = wmma_bf16(a, b0, c0);
  c1 = wmma_bf16(a, b1, c1);
  c2 = wmma_bf16(a, b2, c2);
  c3 = wmma_bf16(a, b3, c3);

  const int n = lane & 15;
  const int rb = (lane >> 4) << 3;
#pragma unroll
  for (int r = 0; r < 8; ++r) {
    const size_t o = ((size_t)h * MTOT + (m0 + rb + r)) * DKV;
    if (outB) {
      outB[o +  0 + n] = (bf16_t)c0[r];
      outB[o + 16 + n] = (bf16_t)c1[r];
      outB[o + 32 + n] = (bf16_t)c2[r];
      outB[o + 48 + n] = (bf16_t)c3[r];
    }
    if (outF) {
      outF[o +  0 + n] = c0[r];
      outF[o + 16 + n] = c1[r];
      outF[o + 32 + n] = c2[r];
      outF[o + 48 + n] = c3[r];
    }
  }
}

// --------------------------------------- fused scores+softmax+top1-mask+AV
// grid = (16, 128=h*B+b), block = 128 (4 waves, each owning one 16-query tile)
__global__ __launch_bounds__(128) void attn_kernel(
    const bf16_t* __restrict__ Qp,   // [H*B][L][64] bf16
    const bf16_t* __restrict__ Kp,   // [H*B][L][64] bf16
    const float*  __restrict__ Vf,   // [H*B][L][64] f32
    float*  __restrict__ attnOut,    // [H*B][L][L] f32  (masked probs)
    bf16_t* __restrict__ headsOut)   // [B][L][H*64] bf16
{
  const float scale = 0.04419417382415922f;  // 1/sqrt(512)
  __shared__ float outbuf[4][16][DKV];
  const int hb   = blockIdx.y;       // h*16 + b
  const int h    = hb >> 4;
  const int b    = hb & 15;
  const int wave = threadIdx.x >> 5;
  const int lane = threadIdx.x & 31;
  const int q0   = (blockIdx.x * 4 + wave) * 16;

  const bf16_t* Qb = Qp + ((size_t)hb * LLEN + q0) * DKV;
  const bf16_t* Kb = Kp + (size_t)hb * LLEN * DKV;
  const float*  Vb = Vf + (size_t)hb * LLEN * DKV;

  v16bf a0 = load_a16x32(Qb + 0,  DKV, lane);
  v16bf a1 = load_a16x32(Qb + 32, DKV, lane);

  for (int i = lane; i < 16 * DKV; i += 32) ((float*)outbuf[wave])[i] = 0.f;

  // ---- pass 1: online (max, sum) per row; each lane sees 1 column per tile
  float mrow[8], zrow[8];
#pragma unroll
  for (int r = 0; r < 8; ++r) { mrow[r] = -__builtin_inff(); zrow[r] = 0.f; }

  {
    v16bf b0 = load_b32x16(Kb + 0,  DKV, lane);
    v16bf b1 = load_b32x16(Kb + 32, DKV, lane);
    for (int kt = 0; kt < 63; ++kt) {
      const bf16_t* Nb = Kb + (size_t)(kt + 1) * 16 * DKV;
      v16bf n0 = load_b32x16(Nb + 0,  DKV, lane);
      v16bf n1 = load_b32x16(Nb + 32, DKV, lane);
      v8f c = {};
      c = wmma_bf16(a0, b0, c);
      c = wmma_bf16(a1, b1, c);
#pragma unroll
      for (int r = 0; r < 8; ++r) {
        float s = c[r];
        if (s > mrow[r]) {
          zrow[r] = zrow[r] * __expf((mrow[r] - s) * scale) + 1.f;
          mrow[r] = s;
        } else {
          zrow[r] += __expf((s - mrow[r]) * scale);
        }
      }
      b0 = n0; b1 = n1;
    }
    v8f c = {};
    c = wmma_bf16(a0, b0, c);
    c = wmma_bf16(a1, b1, c);
#pragma unroll
    for (int r = 0; r < 8; ++r) {
      float s = c[r];
      if (s > mrow[r]) {
        zrow[r] = zrow[r] * __expf((mrow[r] - s) * scale) + 1.f;
        mrow[r] = s;
      } else {
        zrow[r] += __expf((s - mrow[r]) * scale);
      }
    }
  }

  // merge across the 16 lanes that share each row (xor stays inside half-wave)
#pragma unroll
  for (int off = 1; off < 16; off <<= 1) {
#pragma unroll
    for (int r = 0; r < 8; ++r) {
      float mo = __shfl_xor(mrow[r], off, 32);
      float zo = __shfl_xor(zrow[r], off, 32);
      float mn = fmaxf(mrow[r], mo);
      zrow[r] = zrow[r] * __expf((mrow[r] - mn) * scale)
              + zo      * __expf((mo      - mn) * scale);
      mrow[r] = mn;
    }
  }
  float invZ[8];
#pragma unroll
  for (int r = 0; r < 8; ++r) invZ[r] = 1.f / zrow[r];
  // max prob == exp(0)*invZ == invZ exactly (monotone, <=1 factors cannot exceed it)

  // ---- pass 2: recompute scores, write masked probs, sparse-accumulate AV
  const int col16 = lane & 15;
  const int rb    = (lane >> 4) << 3;
  {
    v16bf b0 = load_b32x16(Kb + 0,  DKV, lane);
    v16bf b1 = load_b32x16(Kb + 32, DKV, lane);
    for (int kt = 0; kt < 64; ++kt) {
      v16bf n0, n1;
      if (kt < 63) {
        const bf16_t* Nb = Kb + (size_t)(kt + 1) * 16 * DKV;
        n0 = load_b32x16(Nb + 0,  DKV, lane);
        n1 = load_b32x16(Nb + 32, DKV, lane);
      } else {
        n0 = b0; n1 = b1;
      }
      v8f c = {};
      c = wmma_bf16(a0, b0, c);
      c = wmma_bf16(a1, b1, c);
      const int col = kt * 16 + col16;
#pragma unroll
      for (int r = 0; r < 8; ++r) {
        float p = __expf((c[r] - mrow[r]) * scale) * invZ[r];
        float w = (p == invZ[r]) ? p : 0.f;
        const int row = rb + r;
        attnOut[((size_t)hb * LLEN + (q0 + row)) * LLEN + col] = w;
        if (w != 0.f) {                       // top-1 hit: rare
          const float* vrow = Vb + (size_t)col * DKV;
#pragma unroll
          for (int e = 0; e < DKV; ++e)
            atomicAdd(&outbuf[wave][row][e], w * vrow[e]);
        }
      }
      b0 = n0; b1 = n1;
    }
  }

  // ---- epilogue: heads-concat layout [b][q][h*64+e], bf16 for next GEMM
  for (int i = lane; i < 16 * DKV; i += 32) {
    const int row = i >> 6, e = i & 63;
    headsOut[((size_t)b * LLEN + q0 + row) * (HH * DKV) + h * DKV + e] =
        (bf16_t)outbuf[wave][row][e];
  }
}

// ------------------------------------------- output projection (+bias), bf16 out
// out1[m][d] = sum_e A[m][e] * pw[d][e] + pb[d]   (M=16384, K=512, N=512)
__global__ __launch_bounds__(128) void proj_gemm(
    const bf16_t* __restrict__ A,    // [16384][512]
    const bf16_t* __restrict__ Bw,   // proj_w bf16 [512][512] (row d contiguous in e=k)
    const float*  __restrict__ bias, // [512]
    bf16_t* __restrict__ outB)       // [16384][512]
{
  const int K = DMODEL, N = DMODEL;
  const int wave = threadIdx.x >> 5;
  const int lane = threadIdx.x & 31;
  const int task = blockIdx.x * 4 + wave;   // 8192 tasks
  const int m0 = (task >> 3) * 16;
  const int n0 = (task & 7) * 64;
  const bf16_t* Ab = A + (size_t)m0 * K;
  const bf16_t* Bbase = Bw + (size_t)n0 * K;
  v8f c0 = {}, c1 = {}, c2 = {}, c3 = {};

  v16bf a  = load_a16x32(Ab, K, lane);
  v16bf b0 = load_b32x16(Bbase + (size_t)0  * K, K, lane);
  v16bf b1 = load_b32x16(Bbase + (size_t)16 * K, K, lane);
  v16bf b2 = load_b32x16(Bbase + (size_t)32 * K, K, lane);
  v16bf b3 = load_b32x16(Bbase + (size_t)48 * K, K, lane);
  for (int k0 = 0; k0 < K - 32; k0 += 32) {
    const int kn = k0 + 32;
    v16bf an = load_a16x32(Ab + kn, K, lane);
    v16bf n0v = load_b32x16(Bbase + (size_t)0  * K + kn, K, lane);
    v16bf n1v = load_b32x16(Bbase + (size_t)16 * K + kn, K, lane);
    v16bf n2v = load_b32x16(Bbase + (size_t)32 * K + kn, K, lane);
    v16bf n3v = load_b32x16(Bbase + (size_t)48 * K + kn, K, lane);
    c0 = wmma_bf16(a, b0, c0);
    c1 = wmma_bf16(a, b1, c1);
    c2 = wmma_bf16(a, b2, c2);
    c3 = wmma_bf16(a, b3, c3);
    a = an; b0 = n0v; b1 = n1v; b2 = n2v; b3 = n3v;
  }
  c0 = wmma_bf16(a, b0, c0);
  c1 = wmma_bf16(a, b1, c1);
  c2 = wmma_bf16(a, b2, c2);
  c3 = wmma_bf16(a, b3, c3);

  const int n = lane & 15;
  const int rb = (lane >> 4) << 3;
#pragma unroll
  for (int r = 0; r < 8; ++r) {
    const size_t o = (size_t)(m0 + rb + r) * N + n0;
    outB[o +  0 + n] = (bf16_t)(c0[r] + bias[n0 +  0 + n]);
    outB[o + 16 + n] = (bf16_t)(c1[r] + bias[n0 + 16 + n]);
    outB[o + 32 + n] = (bf16_t)(c2[r] + bias[n0 + 32 + n]);
    outB[o + 48 + n] = (bf16_t)(c3[r] + bias[n0 + 48 + n]);
  }
}

// --------------------- final linear on concat([q, out1]) as two summed GEMMs
// out[m][d] = sum_e q[m][e]*lw[d][e] + sum_e o1[m][e]*lw[d][512+e] + lb[d]
__global__ __launch_bounds__(128) void final_gemm(
    const bf16_t* __restrict__ A0,   // qbf  [16384][512]
    const bf16_t* __restrict__ A1,   // out1 [16384][512]
    const bf16_t* __restrict__ Lw,   // lin_w bf16 [512][1024]
    const float*  __restrict__ bias, // [512]
    float* __restrict__ out)         // [16384][512] f32
{
  const int K = DMODEL, N = DMODEL, LDB = 2 * DMODEL;
  const int wave = threadIdx.x >> 5;
  const int lane = threadIdx.x & 31;
  const int task = blockIdx.x * 4 + wave;   // 8192 tasks
  const int m0 = (task >> 3) * 16;
  const int n0 = (task & 7) * 64;
  v8f c0 = {}, c1 = {}, c2 = {}, c3 = {};

#pragma unroll
  for (int half = 0; half < 2; ++half) {
    const bf16_t* Ab = (half == 0 ? A0 : A1) + (size_t)m0 * K;
    const bf16_t* Bbase = Lw + (size_t)n0 * LDB + half * DMODEL;
    v16bf a  = load_a16x32(Ab, K, lane);
    v16bf b0 = load_b32x16(Bbase + (size_t)0  * LDB, LDB, lane);
    v16bf b1 = load_b32x16(Bbase + (size_t)16 * LDB, LDB, lane);
    v16bf b2 = load_b32x16(Bbase + (size_t)32 * LDB, LDB, lane);
    v16bf b3 = load_b32x16(Bbase + (size_t)48 * LDB, LDB, lane);
    for (int k0 = 0; k0 < K - 32; k0 += 32) {
      const int kn = k0 + 32;
      v16bf an = load_a16x32(Ab + kn, K, lane);
      v16bf n0v = load_b32x16(Bbase + (size_t)0  * LDB + kn, LDB, lane);
      v16bf n1v = load_b32x16(Bbase + (size_t)16 * LDB + kn, LDB, lane);
      v16bf n2v = load_b32x16(Bbase + (size_t)32 * LDB + kn, LDB, lane);
      v16bf n3v = load_b32x16(Bbase + (size_t)48 * LDB + kn, LDB, lane);
      c0 = wmma_bf16(a, b0, c0);
      c1 = wmma_bf16(a, b1, c1);
      c2 = wmma_bf16(a, b2, c2);
      c3 = wmma_bf16(a, b3, c3);
      a = an; b0 = n0v; b1 = n1v; b2 = n2v; b3 = n3v;
    }
    c0 = wmma_bf16(a, b0, c0);
    c1 = wmma_bf16(a, b1, c1);
    c2 = wmma_bf16(a, b2, c2);
    c3 = wmma_bf16(a, b3, c3);
  }

  const int n = lane & 15;
  const int rb = (lane >> 4) << 3;
#pragma unroll
  for (int r = 0; r < 8; ++r) {
    const size_t o = (size_t)(m0 + rb + r) * N + n0;
    out[o +  0 + n] = c0[r] + bias[n0 +  0 + n];
    out[o + 16 + n] = c1[r] + bias[n0 + 16 + n];
    out[o + 32 + n] = c2[r] + bias[n0 + 32 + n];
    out[o + 48 + n] = c3[r] + bias[n0 + 48 + n];
  }
}

// ---------------------------------------------------------------------------
extern "C" void kernel_launch(void* const* d_in, const int* in_sizes, int n_in,
                              void* d_out, int out_size, void* d_ws, size_t ws_size,
                              hipStream_t stream) {
  const float* q_data = (const float*)d_in[0];
  const float* k_data = (const float*)d_in[1];
  const float* v_data = (const float*)d_in[2];
  const float* w_qs   = (const float*)d_in[3];
  const float* w_ks   = (const float*)d_in[4];
  const float* w_vs   = (const float*)d_in[5];
  const float* proj_w = (const float*)d_in[6];
  const float* proj_b = (const float*)d_in[7];
  const float* lin_w  = (const float*)d_in[8];
  const float* lin_b  = (const float*)d_in[9];

  float* outp    = (float*)d_out;                       // [16,1024,512]
  float* attnOut = outp + (size_t)MTOT * DMODEL;        // [128,1024,1024]

  size_t off = 0;
  auto take = [&](size_t bytes) -> void* {
    void* p = (char*)d_ws + off;
    off += (bytes + 255) & ~(size_t)255;
    return p;
  };
  bf16_t* qbf  = (bf16_t*)take((size_t)MTOT * DMODEL * 2);
  bf16_t* kbf  = (bf16_t*)take((size_t)MTOT * DMODEL * 2);   // reused as headsOut
  bf16_t* vbf  = (bf16_t*)take((size_t)MTOT * DMODEL * 2);   // reused as out1
  bf16_t* wqt  = (bf16_t*)take((size_t)HH * DMODEL * DKV * 2);
  bf16_t* wkt  = (bf16_t*)take((size_t)HH * DMODEL * DKV * 2);
  bf16_t* wvt  = (bf16_t*)take((size_t)HH * DMODEL * DKV * 2);
  bf16_t* pwbf = (bf16_t*)take((size_t)DMODEL * DMODEL * 2);
  bf16_t* lwbf = (bf16_t*)take((size_t)DMODEL * 2 * DMODEL * 2);
  bf16_t* Qp   = (bf16_t*)take((size_t)HH * MTOT * DKV * 2);
  bf16_t* Kp   = (bf16_t*)take((size_t)HH * MTOT * DKV * 2);
  float*  Vfp  = (float*) take((size_t)HH * MTOT * DKV * 4);
  bf16_t* headsOut = kbf;   // k_data bf16 no longer needed after QKV GEMMs
  bf16_t* out1     = vbf;   // v_data bf16 no longer needed after QKV GEMMs

  const int nX = MTOT * DMODEL;
  cvt_f32_bf16<<<2048, 256, 0, stream>>>(q_data, qbf, nX);
  cvt_f32_bf16<<<2048, 256, 0, stream>>>(k_data, kbf, nX);
  cvt_f32_bf16<<<2048, 256, 0, stream>>>(v_data, vbf, nX);
  cvt_transpose_w<<<256, 256, 0, stream>>>(w_qs, wqt);
  cvt_transpose_w<<<256, 256, 0, stream>>>(w_ks, wkt);
  cvt_transpose_w<<<256, 256, 0, stream>>>(w_vs, wvt);
  cvt_f32_bf16<<<512, 256, 0, stream>>>(proj_w, pwbf, DMODEL * DMODEL);
  cvt_f32_bf16<<<1024, 256, 0, stream>>>(lin_w, lwbf, DMODEL * 2 * DMODEL);

  qkv_gemm<<<dim3(256, 8), 128, 0, stream>>>(qbf, wqt, Qp, nullptr);
  qkv_gemm<<<dim3(256, 8), 128, 0, stream>>>(kbf, wkt, Kp, nullptr);
  qkv_gemm<<<dim3(256, 8), 128, 0, stream>>>(vbf, wvt, nullptr, Vfp);

  attn_kernel<<<dim3(16, 128), 128, 0, stream>>>(Qp, Kp, Vfp, attnOut, headsOut);

  proj_gemm<<<2048, 128, 0, stream>>>(headsOut, pwbf, proj_b, out1);
  final_gemm<<<2048, 128, 0, stream>>>(qbf, out1, lwbf, lin_b, outp);
}
